// BiLSTM_CRF_1726576856660
// MI455X (gfx1250) — matrix operand order repaired and checked by reading, not compiled, not run on hardware
//
#include <hip/hip_runtime.h>

// Problem constants (match reference)
#define L_    2048
#define E_    512
#define H2_   512
#define G_    2048          // 4*H2 gates
#define K_    11
#define START_ 9
#define STOP_  10
#define NEG_  (-10000.0f)
#define NWG_DIR 64          // persistent workgroups per direction

typedef __bf16 bf16;
typedef __attribute__((ext_vector_type(16))) __bf16 v16bf;
typedef __attribute__((ext_vector_type(8)))  __bf16 v8bf;
typedef __attribute__((ext_vector_type(8)))  float  v8f;
typedef __attribute__((ext_vector_type(4)))  int    v4i;
typedef __attribute__((ext_vector_type(4)))  unsigned int u32x4;
typedef __attribute__((ext_vector_type(8)))  unsigned int u32x8;

// LDS byte offset of a __shared__ object (generic -> AS3 -> int)
__device__ __forceinline__ unsigned lds_offset_of(const void* p) {
  return (unsigned)(size_t)(const __attribute__((address_space(3))) void*)p;
}

// ---------------------------------------------------------------------------
// 1) Embedding gather + f32->bf16 conversion of x
// ---------------------------------------------------------------------------
__global__ __launch_bounds__(256) void gather_embed_kernel(
    const int* __restrict__ sent, const float* __restrict__ embed,
    bf16* __restrict__ xbf) {
  const int t = blockIdx.x;
  const float* src = embed + (size_t)sent[t] * E_;
  bf16* dst = xbf + (size_t)t * E_;
  for (int e = threadIdx.x; e < E_; e += 256) dst[e] = (bf16)src[e];
}

__global__ __launch_bounds__(256) void cvt_bf16_kernel(
    const float* __restrict__ src, bf16* __restrict__ dst, int n) {
  int i = blockIdx.x * 256 + threadIdx.x;
  if (i < n) dst[i] = (bf16)src[i];
}

// ---------------------------------------------------------------------------
// 2) xg[dir][L][G] = x_bf @ Wih_dir^T + (bih+bhh)   — WMMA bf16, f32 acc.
//    One wave computes a 16(M) x 64(N) strip: 4 accumulators, A fragment
//    reused 4x per K-step. K loop = 512/32 = 16 steps -> 64 WMMAs per wave.
// ---------------------------------------------------------------------------
__global__ __launch_bounds__(256) void xg_gemm_kernel(
    const bf16* __restrict__ xbf, const bf16* __restrict__ wih_bf,
    const float* __restrict__ bih_f, const float* __restrict__ bhh_f,
    const float* __restrict__ bih_b, const float* __restrict__ bhh_b,
    float* __restrict__ xg) {
  const int wave = (blockIdx.x * 256 + threadIdx.x) >> 5;   // 8192 waves
  const int lane = threadIdx.x & 31;
  const int l15  = lane & 15;
  const int tn4   = wave & 31;          // which 64-column strip
  const int tileM = (wave >> 5) & 127;
  const int dir   = wave >> 12;

  const bf16* A0 = xbf + (size_t)(tileM * 16 + l15) * E_;
  const bf16* B0 = wih_bf + (size_t)dir * G_ * E_
                          + (size_t)(tn4 * 64 + l15) * E_;
  const int aoff = (lane & 16) ? 16 : 0;   // A k order used here (contiguous): K 0..15 / 16..31
  const int boff = (lane & 16) ? 16 : 0;   // B k order used here (contiguous): K 0..15 / 16..31

  v8f acc0 = {}, acc1 = {}, acc2 = {}, acc3 = {};
  for (int kk = 0; kk < E_; kk += 32) {
    v8bf alo = *(const v8bf*)(A0 + kk + aoff);
    v8bf ahi = *(const v8bf*)(A0 + kk + aoff + 8);
    v16bf a;
#pragma unroll
    for (int i = 0; i < 8; ++i) { a[i] = alo[i]; a[8 + i] = ahi[i]; }
    v16bf b0 = *(const v16bf*)(B0 + (size_t)0 * 16 * E_ + kk + boff);
    v16bf b1 = *(const v16bf*)(B0 + (size_t)1 * 16 * E_ + kk + boff);
    v16bf b2 = *(const v16bf*)(B0 + (size_t)2 * 16 * E_ + kk + boff);
    v16bf b3 = *(const v16bf*)(B0 + (size_t)3 * 16 * E_ + kk + boff);
    acc0 = __builtin_amdgcn_wmma_f32_16x16x32_bf16(false, a, false, b0,
                                                   (short)0, acc0, false, false);
    acc1 = __builtin_amdgcn_wmma_f32_16x16x32_bf16(false, a, false, b1,
                                                   (short)0, acc1, false, false);
    acc2 = __builtin_amdgcn_wmma_f32_16x16x32_bf16(false, a, false, b2,
                                                   (short)0, acc2, false, false);
    acc3 = __builtin_amdgcn_wmma_f32_16x16x32_bf16(false, a, false, b3,
                                                   (short)0, acc3, false, false);
  }

  const int mrow = tileM * 16 + ((lane >> 4) << 3);
  float* obase = xg + (size_t)dir * L_ * G_ + (size_t)mrow * G_;
#define STORE_STRIP(RB, ACC)                                                 \
  {                                                                          \
    const int n = tn4 * 64 + (RB) * 16 + l15;                                \
    const float bias = dir ? (bih_b[n] + bhh_b[n]) : (bih_f[n] + bhh_f[n]);  \
    _Pragma("unroll")                                                        \
    for (int r = 0; r < 8; ++r) obase[(size_t)r * G_ + n] = ACC[r] + bias;   \
  }
  STORE_STRIP(0, acc0)
  STORE_STRIP(1, acc1)
  STORE_STRIP(2, acc2)
  STORE_STRIP(3, acc3)
#undef STORE_STRIP
}

// ---------------------------------------------------------------------------
// 3) Persistent LSTM recurrence. 128 blocks = 2 dirs x 64 groups.
//    Whh slice in VGPRs; h_prev staged into LDS by the Tensor Data Mover
//    (wave 0 builds a 1-D D# in SGPRs, tensor_load_to_lds, s_wait_tensorcnt).
// ---------------------------------------------------------------------------
__global__ __launch_bounds__(256) void lstm_recurrent_kernel(
    const bf16* __restrict__ whh_bf, const float* __restrict__ xg,
    const float* __restrict__ h0, const float* __restrict__ c0,
    float* __restrict__ hs, int* __restrict__ done) {
  __shared__ __align__(16) float h_sh[H2_];
  __shared__ float g_sh[32];

  const int dir = blockIdx.x >> 6;
  const int grp = blockIdx.x & 63;
  const int tid = threadIdx.x;
  const int j0  = grp * 8;

  const int r   = tid >> 3;        // 0..31 : owned Whh row
  const int oct = tid & 7;         // 0..7  : 64-elem chunk within row
  const int q   = r >> 3;          // gate (i,f,g,o)
  const int jj  = r & 7;
  const int grow = q * H2_ + j0 + jj;

  // Whh slice -> registers (read once)
  float wreg[64];
  {
    const v8bf* wp = (const v8bf*)(whh_bf + (size_t)dir * G_ * H2_
                                   + (size_t)grow * H2_ + oct * 64);
#pragma unroll
    for (int c = 0; c < 8; ++c) {
      v8bf v = wp[c];
#pragma unroll
      for (int u = 0; u < 8; ++u) wreg[c * 8 + u] = (float)v[u];
    }
  }

  float c_reg = (tid < 8) ? c0[dir * H2_ + j0 + tid] : 0.0f;

  const float* xgd = xg + (size_t)dir * L_ * G_;
  float* hsd = hs + (size_t)dir * L_ * H2_;
  int* dcnt = done + dir * L_;
  const unsigned h_lds = lds_offset_of(&h_sh[0]);

  for (int s = 0; s < L_; ++s) {
    const int t = dir ? (L_ - 1 - s) : s;
    const float* hprev = (s == 0) ? (h0 + dir * H2_)
                                  : (hsd + (size_t)(dir ? (t + 1) : (t - 1)) * H2_);

    // --- TDM: DMA 512 f32 (2 KB) h_prev -> LDS, wave 0 only ---------------
    if (tid < 32) {
      unsigned long long ga = (unsigned long long)(size_t)hprev;
      u32x4 g0;
      g0[0] = 1u;                                   // count=1, user mode
      g0[1] = h_lds;                                // lds_addr
      g0[2] = (unsigned)ga;                         // global_addr[31:0]
      g0[3] = (unsigned)((ga >> 32) & 0x1ffffffu)   // global_addr[56:32]
            | (2u << 30);                           // type=2 ("image")
      u32x8 g1;
      g1[0] = (2u << 16);                           // data_size=4B, mask=0
      g1[1] = ((unsigned)H2_ & 0xffffu) << 16;      // tensor_dim0 lo16
      g1[2] = ((unsigned)H2_ >> 16) | (1u << 16);   // dim0 hi16 | tensor_dim1=1
      g1[3] = ((unsigned)H2_) << 16;                // dim1 hi=0 | tile_dim0=512
      g1[4] = 1u;                                   // tile_dim1=1, tile_dim2=0
      g1[5] = (unsigned)H2_;                        // tensor_dim0_stride lo32
      g1[6] = 0u;                                   // stride hi | dim1_stride lo
      g1[7] = 0u;
      asm volatile("tensor_load_to_lds %0, %1" :: "s"(g0), "s"(g1) : "memory");
      __builtin_amdgcn_s_wait_tensorcnt(0);
    }
    __syncthreads();

    // partial dot: 64 fmas from registers x LDS
    float sum = 0.0f;
    const float4* hv = (const float4*)(h_sh + oct * 64);
#pragma unroll
    for (int c = 0; c < 16; ++c) {
      float4 h4 = hv[c];
      sum = __builtin_fmaf(wreg[4 * c + 0], h4.x, sum);
      sum = __builtin_fmaf(wreg[4 * c + 1], h4.y, sum);
      sum = __builtin_fmaf(wreg[4 * c + 2], h4.z, sum);
      sum = __builtin_fmaf(wreg[4 * c + 3], h4.w, sum);
    }
    sum += __shfl_xor(sum, 1);
    sum += __shfl_xor(sum, 2);
    sum += __shfl_xor(sum, 4);
    if (oct == 0) g_sh[r] = sum;
    __syncthreads();

    if (tid < 8) {
      const int j = j0 + tid;
      const float* xr = xgd + (size_t)t * G_;
      float gi = xr[j]            + g_sh[tid];
      float gf = xr[H2_ + j]      + g_sh[8 + tid];
      float gg = xr[2 * H2_ + j]  + g_sh[16 + tid];
      float go = xr[3 * H2_ + j]  + g_sh[24 + tid];
      float si = 1.0f / (1.0f + __expf(-gi));
      float sf = 1.0f / (1.0f + __expf(-gf));
      float so = 1.0f / (1.0f + __expf(-go));
      c_reg = sf * c_reg + si * tanhf(gg);
      hsd[(size_t)t * H2_ + j] = so * tanhf(c_reg);
    }

    // prefetch next step's xg gate rows (global_prefetch_b8)
    if (s + 1 < L_ && tid < 4) {
      const int tn = dir ? (t - 1) : (t + 1);
      __builtin_prefetch(xgd + (size_t)tn * G_ + tid * H2_ + j0, 0, 1);
    }

    // cross-workgroup step barrier (cluster barrier is NOP when unclustered)
    __builtin_amdgcn_s_cluster_barrier();
    __threadfence();
    __syncthreads();
    if (tid == 0) {
      __hip_atomic_fetch_add(&dcnt[s], 1, __ATOMIC_RELEASE,
                             __HIP_MEMORY_SCOPE_AGENT);
      while (__hip_atomic_load(&dcnt[s], __ATOMIC_ACQUIRE,
                               __HIP_MEMORY_SCOPE_AGENT) < NWG_DIR)
        __builtin_amdgcn_s_sleep(1);
    }
    __syncthreads();
  }
}

// ---------------------------------------------------------------------------
// 4) feats[t][k] = [hf(t) ; hb(t)] . W_tag[k] + b_tag[k]  (async LDS staging)
// ---------------------------------------------------------------------------
__global__ __launch_bounds__(256) void feats_kernel(
    const float* __restrict__ hf, const float* __restrict__ hb,
    const float* __restrict__ Wtag, const float* __restrict__ btag,
    float* __restrict__ feats) {
  __shared__ __align__(16) float vsh[2 * H2_];
  __shared__ float red[8];
  const int t = blockIdx.x, tid = threadIdx.x;

#if __has_builtin(__builtin_amdgcn_global_load_async_to_lds_b128)
  {
    const float* src = (tid < 128) ? (hf + (size_t)t * H2_ + tid * 4)
                                   : (hb + (size_t)t * H2_ + (tid - 128) * 4);
    float* dst = (tid < 128) ? &vsh[tid * 4] : &vsh[H2_ + (tid - 128) * 4];
    __builtin_amdgcn_global_load_async_to_lds_b128(
        (__attribute__((address_space(1))) v4i*)(size_t)src,
        (__attribute__((address_space(3))) v4i*)dst, 0, 0);
  }
#if __has_builtin(__builtin_amdgcn_s_wait_asynccnt)
  __builtin_amdgcn_s_wait_asynccnt(0);
#else
  asm volatile("s_wait_asynccnt 0x0" ::: "memory");
#endif
#else
  for (int i = tid; i < H2_; i += 256) {
    vsh[i]       = hf[(size_t)t * H2_ + i];
    vsh[H2_ + i] = hb[(size_t)t * H2_ + i];
  }
#endif
  __syncthreads();

  for (int k = 0; k < K_; ++k) {
    float s = 0.0f;
    for (int i = tid; i < 2 * H2_; i += 256) s += vsh[i] * Wtag[k * 2 * H2_ + i];
    for (int m = 16; m > 0; m >>= 1) s += __shfl_down(s, m);
    if ((tid & 31) == 0) red[tid >> 5] = s;
    __syncthreads();
    if (tid == 0) {
      float tot = 0.0f;
      for (int w = 0; w < 8; ++w) tot += red[w];
      feats[t * 16 + k] = tot + btag[k];
    }
    __syncthreads();
  }
}

// ---------------------------------------------------------------------------
// 5) Viterbi DP + backtrace. Single wave; backpointers in LDS (22.5 KB).
// ---------------------------------------------------------------------------
__global__ __launch_bounds__(32) void viterbi_kernel(
    const float* __restrict__ feats, const float* __restrict__ trans,
    float* __restrict__ out) {
  __shared__ float fv[16];
  __shared__ float nfv[16];
  __shared__ unsigned char bp[L_ * K_];
  const int to = threadIdx.x;
  if (to < K_) fv[to] = (to == START_) ? 0.0f : NEG_;
  __syncthreads();
  for (int t = 0; t < L_; ++t) {
    if (to < K_) {
      float best = -3.4e38f; int bi = 0;
      for (int f = 0; f < K_; ++f) {
        float v = fv[f] + trans[to * K_ + f];
        if (v > best) { best = v; bi = f; }
      }
      nfv[to] = best + feats[t * 16 + to];
      bp[t * K_ + to] = (unsigned char)bi;
    }
    __syncthreads();
    if (to < K_) fv[to] = nfv[to];
    __syncthreads();
  }
  if (to == 0) {
    float best = -3.4e38f; int bi = 0;
    for (int j = 0; j < K_; ++j) {
      float v = fv[j] + trans[STOP_ * K_ + j];
      if (j == STOP_ || j == START_) v = NEG_;
      if (v > best) { best = v; bi = j; }
    }
    out[0] = best;
    out[1 + (L_ - 1)] = (float)bi;
    int cur = bi;
    for (int t = L_ - 1; t >= 1; --t) {
      cur = (int)bp[t * K_ + cur];
      out[1 + (t - 1)] = (float)cur;
    }
  }
}

// ---------------------------------------------------------------------------
// Host launcher
// ---------------------------------------------------------------------------
extern "C" void kernel_launch(void* const* d_in, const int* in_sizes, int n_in,
                              void* d_out, int out_size, void* d_ws, size_t ws_size,
                              hipStream_t stream) {
  const int*   sentence = (const int*)  d_in[0];
  const float* embed    = (const float*)d_in[1];
  const float* Wih_f    = (const float*)d_in[2];
  const float* Whh_f    = (const float*)d_in[3];
  const float* bih_f    = (const float*)d_in[4];
  const float* bhh_f    = (const float*)d_in[5];
  const float* Wih_b    = (const float*)d_in[6];
  const float* Whh_b    = (const float*)d_in[7];
  const float* bih_b    = (const float*)d_in[8];
  const float* bhh_b    = (const float*)d_in[9];
  const float* h0       = (const float*)d_in[10];
  const float* c0       = (const float*)d_in[11];
  const float* W_tag    = (const float*)d_in[12];
  const float* b_tag    = (const float*)d_in[13];
  const float* trans    = (const float*)d_in[14];
  float* out = (float*)d_out;

  char* ws = (char*)d_ws;
  size_t off = 0;
  auto alloc = [&](size_t bytes) {
    size_t r = off;
    off = (off + bytes + 255) & ~(size_t)255;
    return r;
  };
  int*  done    = (int*) (ws + alloc(2 * L_ * sizeof(int)));
  bf16* xbf     = (bf16*)(ws + alloc((size_t)L_ * E_ * sizeof(bf16)));
  bf16* wih_bf  = (bf16*)(ws + alloc((size_t)2 * G_ * E_ * sizeof(bf16)));
  bf16* whh_bf  = (bf16*)(ws + alloc((size_t)2 * G_ * H2_ * sizeof(bf16)));
  float* xg     = (float*)(ws + alloc((size_t)2 * L_ * G_ * sizeof(float)));
  float* hs     = (float*)(ws + alloc((size_t)2 * L_ * H2_ * sizeof(float)));
  float* feats  = (float*)(ws + alloc((size_t)L_ * 16 * sizeof(float)));
  (void)ws_size; (void)n_in; (void)in_sizes; (void)out_size;

  (void)hipMemsetAsync(done, 0, 2 * L_ * sizeof(int), stream);

  gather_embed_kernel<<<L_, 256, 0, stream>>>(sentence, embed, xbf);

  const int NW = G_ * E_;
  cvt_bf16_kernel<<<(NW + 255) / 256, 256, 0, stream>>>(Wih_f, wih_bf,      NW);
  cvt_bf16_kernel<<<(NW + 255) / 256, 256, 0, stream>>>(Wih_b, wih_bf + NW, NW);
  cvt_bf16_kernel<<<(NW + 255) / 256, 256, 0, stream>>>(Whh_f, whh_bf,      NW);
  cvt_bf16_kernel<<<(NW + 255) / 256, 256, 0, stream>>>(Whh_b, whh_bf + NW, NW);

  // 2 dirs * 128 M-tiles * 32 N-strips, 8 waves/block -> 1024 blocks
  xg_gemm_kernel<<<1024, 256, 0, stream>>>(xbf, wih_bf, bih_f, bhh_f,
                                           bih_b, bhh_b, xg);

  lstm_recurrent_kernel<<<2 * NWG_DIR, 256, 0, stream>>>(whh_bf, xg, h0, c0,
                                                         hs, done);

  feats_kernel<<<L_, 256, 0, stream>>>(hs, hs + (size_t)L_ * H2_, W_tag, b_tag,
                                       feats);

  viterbi_kernel<<<1, 32, 0, stream>>>(feats, trans, out);
}
